// KimiSparseMoE_43963285242609
// MI455X (gfx1250) — compile-verified, matrix operand order checked by reading, not run on hardware
//
#include <hip/hip_runtime.h>

// ---------------------------------------------------------------------------
// Kimi sparse MoE forward for MI455X (gfx1250, wave32, WMMA, async-to-LDS).
//   D=2048, E=32, DFF=1024, T=1024, TOP_K=4, scaling 2.5, 1 shared expert.
//
// Roofline: 830 MB fp32 weights -> 36us HBM floor at 23.3 TB/s; needs
// ~1.8 PFLOP/s matmul to stay memory-bound, so weights are converted to bf16
// in-flight and fed to v_wmma_f32_16x16x32_bf16 (fp32 accumulate).
// Token-gather computes only the top-4 assignments (8x FLOP saving vs the
// reference's dense expert scan). TILE_M=128 token tiles amortize each
// expert's 24MB weight stream to ~1 HBM pass.
//
// LDS (320KB/WGP): 256KB bf16 H tile + double-buffered weight staging:
// weight K-slabs are copied global->LDS with GLOBAL_LOAD_ASYNC_TO_LDS_B128
// (ASYNCcnt-tracked, no VGPR round trip), converted fp32->bf16 cooperatively
// ONCE per block (8x less cvt VALU + 8x less L2 traffic than per-wave
// conversion), then consumed as B-fragments via 128-bit ds loads.
// ---------------------------------------------------------------------------

#define Dh    2048
#define Eh    32
#define DFFh  1024
#define Th    1024
#define TOPKh 4
#define TMh   128      // token tile (8 waves x 16-row M-subtiles)
#define KSLAB 32       // K depth staged per step (= one WMMA K step)

// Dynamic LDS layout (bytes) -- fixed regardless of async availability so the
// host launch size matches the device view.
#define HS_BYTES  (TMh * DFFh * 2)            // 262144: H tile, bf16
#define F_BYTES   (2 * 2 * 64 * KSLAB * 4)    //  32768: fp32 stage, 2 bufs
#define B_BYTES   (2 * 2 * 64 * KSLAB * 2)    //  16384: bf16 stage, 2 bufs
#define LDS_TOTAL (HS_BYTES + F_BYTES + B_BYTES)   // 311296 <= 320K

#if defined(__HIP_DEVICE_COMPILE__) && \
    __has_builtin(__builtin_amdgcn_global_load_async_to_lds_b128) && \
    __has_builtin(__builtin_amdgcn_s_wait_asynccnt)
#define USE_ASYNC 1
#else
#define USE_ASYNC 0
#endif

typedef __bf16 v16bf __attribute__((ext_vector_type(16)));
typedef float  v8f   __attribute__((ext_vector_type(8)));
typedef float  f32x4 __attribute__((ext_vector_type(4)));
typedef unsigned int u32x4 __attribute__((ext_vector_type(4)));

union FragBF  { v16bf v; __bf16 e[16]; u32x4 q[2]; };
union FragAcc { v8f   v; float  f[8]; };

// Fast sigmoid: v_rcp_f32 instead of the IEEE division sequence (~1 ulp,
// invisible after the bf16 round-trip).
__device__ __forceinline__ float fast_sigmoid(float g) {
  return __builtin_amdgcn_rcpf(1.0f + __expf(-g));
}

// Build a bf16 WMMA fragment from two 8-float contiguous runs (the two K-runs
// a lane owns per ISA 7.12.2: K = half*8 + [0..7] and K = 16 + half*8 + [0..7]).
__device__ __forceinline__ v16bf make_bfrag(const float* __restrict__ p0,
                                            const float* __restrict__ p1) {
  f32x4 a0 = *(const f32x4*)(p0);
  f32x4 a1 = *(const f32x4*)(p0 + 4);
  f32x4 b0 = *(const f32x4*)(p1);
  f32x4 b1 = *(const f32x4*)(p1 + 4);
  FragBF f;
#pragma unroll
  for (int i = 0; i < 4; ++i) {
    f.e[i]      = (__bf16)a0[i];
    f.e[4 + i]  = (__bf16)a1[i];
    f.e[8 + i]  = (__bf16)b0[i];
    f.e[12 + i] = (__bf16)b1[i];
  }
  return f.v;
}

// B-fragment from the bf16 LDS stage: two ds_load_b128 per lane.
__device__ __forceinline__ v16bf bfrag_stage(const __bf16* __restrict__ B,
                                             int mat, int row, int half) {
  const __bf16* p = B + (size_t)(mat * 64 + row) * KSLAB;
  FragBF f;
  f.q[0] = *(const u32x4*)(p + half * 8);
  f.q[1] = *(const u32x4*)(p + 16 + half * 8);
  return f.v;
}

#if USE_ASYNC
// The builtin's params are GCC-style int 4-vectors in AS1 (global) / AS3 (LDS):
//   void __builtin_amdgcn_global_load_async_to_lds_b128(v4i AS1*, v4i AS3*,
//                                                       imm offset, imm cpol)
typedef int gv4i __attribute__((vector_size(16)));
#define AS1P(p) ((__attribute__((address_space(1))) gv4i*)(unsigned long long)(p))
#define AS3P(p) ((__attribute__((address_space(3))) gv4i*)(unsigned int)(unsigned long long)(p))
// One 16-byte lane-chunk, global -> LDS, tracked by ASYNCcnt.
__device__ __forceinline__ void async_cp16(const float* g, float* l) {
  __builtin_amdgcn_global_load_async_to_lds_b128(AS1P(g), AS3P(l), 0, 0);
}
#endif

// ---- weight slab staging ---------------------------------------------------
// Slab = NMATS matrices x 64 rows x KSLAB(=32) fp32 K-values.

#if USE_ASYNC
template <int NMATS>
__device__ __forceinline__ void stage_copy_async(float* __restrict__ Fbuf,
                                                 const float* __restrict__ src0,
                                                 const float* __restrict__ src1,
                                                 int rowStride, int kb, int tid) {
  // 16B chunks: NMATS*64*8 of them, spread over 256 threads.
#pragma unroll
  for (int i = tid; i < NMATS * 64 * 8; i += 256) {
    const int mat = i >> 9;          // 512 chunks per matrix
    const int rem = i & 511;
    const int row = rem >> 3;
    const int c   = rem & 7;
    const float* g = (mat ? src1 : src0) + (size_t)row * rowStride + kb + c * 4;
    float* l = Fbuf + (size_t)(mat * 64 + row) * KSLAB + c * 4;
    async_cp16(g, l);
  }
}

template <int NMATS>
__device__ __forceinline__ void stage_convert(const float* __restrict__ Fbuf,
                                              __bf16* __restrict__ Bbuf, int tid) {
  // 16-float segments: NMATS*64*2 of them.
#pragma unroll
  for (int i = tid; i < NMATS * 64 * 2; i += 256) {
    const int mat = i >> 7;
    const int rem = i & 127;
    const int row = rem >> 1;
    const int seg = rem & 1;
    const float* f = Fbuf + (size_t)(mat * 64 + row) * KSLAB + seg * 16;
    __bf16* b = Bbuf + (size_t)(mat * 64 + row) * KSLAB + seg * 16;
    f32x4 v0 = *(const f32x4*)(f);
    f32x4 v1 = *(const f32x4*)(f + 4);
    f32x4 v2 = *(const f32x4*)(f + 8);
    f32x4 v3 = *(const f32x4*)(f + 12);
    FragBF t;
#pragma unroll
    for (int j = 0; j < 4; ++j) {
      t.e[j] = (__bf16)v0[j]; t.e[4 + j] = (__bf16)v1[j];
      t.e[8 + j] = (__bf16)v2[j]; t.e[12 + j] = (__bf16)v3[j];
    }
    *(u32x4*)(b) = t.q[0];
    *(u32x4*)(b + 8) = t.q[1];
  }
}
#else
template <int NMATS>
__device__ __forceinline__ void stage_fill_sync(__bf16* __restrict__ Bbuf,
                                                const float* __restrict__ src0,
                                                const float* __restrict__ src1,
                                                int rowStride, int kb, int tid) {
#pragma unroll
  for (int i = tid; i < NMATS * 64 * 2; i += 256) {
    const int mat = i >> 7;
    const int rem = i & 127;
    const int row = rem >> 1;
    const int seg = rem & 1;
    const float* g = (mat ? src1 : src0) + (size_t)row * rowStride + kb + seg * 16;
    __bf16* b = Bbuf + (size_t)(mat * 64 + row) * KSLAB + seg * 16;
    f32x4 v0 = *(const f32x4*)(g);
    f32x4 v1 = *(const f32x4*)(g + 4);
    f32x4 v2 = *(const f32x4*)(g + 8);
    f32x4 v3 = *(const f32x4*)(g + 12);
    FragBF t;
#pragma unroll
    for (int j = 0; j < 4; ++j) {
      t.e[j] = (__bf16)v0[j]; t.e[4 + j] = (__bf16)v1[j];
      t.e[8 + j] = (__bf16)v2[j]; t.e[12 + j] = (__bf16)v3[j];
    }
    *(u32x4*)(b) = t.q[0];
    *(u32x4*)(b + 8) = t.q[1];
  }
}
#endif

// ---------------------------------------------------------------------------
// Router: gates = x @ gate_w^T, sigmoid, grouped top-k with the reference's
// faithful-to-torch masking quirk. One block per token (gate_w stays in L2).
// ---------------------------------------------------------------------------
__global__ __launch_bounds__(256)
void router_kernel(const float* __restrict__ x,
                   const float* __restrict__ gate_w,
                   const float* __restrict__ bias,
                   int* __restrict__ inds,
                   float* __restrict__ wts) {
  __shared__ float s_red[8][Eh];
  __shared__ float s_sig[Eh];
  const int t = blockIdx.x;
  const int tid = threadIdx.x;
  const int e = tid & 31;
  const int chunk = tid >> 5;

  const float* xr = x + (size_t)t * Dh + chunk * 256;
  const float* gw = gate_w + (size_t)e * Dh + chunk * 256;
  float s = 0.f;
#pragma unroll 4
  for (int i = 0; i < 64; ++i) {
    f32x4 a = *(const f32x4*)(xr + i * 4);
    f32x4 b = *(const f32x4*)(gw + i * 4);
    s += a[0]*b[0] + a[1]*b[1] + a[2]*b[2] + a[3]*b[3];
  }
  s_red[chunk][e] = s;
  __syncthreads();

  if (tid < Eh) {
    float g = 0.f;
#pragma unroll
    for (int c = 0; c < 8; ++c) g += s_red[c][tid];
    s_sig[tid] = fast_sigmoid(g);
  }
  __syncthreads();

  if (tid == 0) {
    float orig[Eh], sb[Eh];
    for (int i = 0; i < Eh; ++i) { orig[i] = s_sig[i]; sb[i] = orig[i] + bias[i]; }
    float gs[4];
    for (int g = 0; g < 4; ++g) {
      float m1 = -1e30f, m2 = -1e30f;
      for (int j = 0; j < 8; ++j) {
        float v = sb[g * 8 + j];
        if (v > m1) { m2 = m1; m1 = v; } else if (v > m2) { m2 = v; }
      }
      gs[g] = m1 + m2;
    }
    int g1 = 0;
    for (int g = 1; g < 4; ++g) if (gs[g] > gs[g1]) g1 = g;
    int g2 = -1;
    for (int g = 0; g < 4; ++g) if (g != g1 && (g2 < 0 || gs[g] > gs[g2])) g2 = g;
    // faithful mask quirk: only column 0 of the 2 winning groups survives
    float ms[Eh];
    for (int i = 0; i < Eh; ++i) ms[i] = 0.f;
    ms[8 * g1] = sb[8 * g1];
    ms[8 * g2] = sb[8 * g2];
    bool taken[Eh] = {};
    int ind[TOPKh]; float wv[TOPKh]; float wsum = 0.f;
    for (int j = 0; j < TOPKh; ++j) {
      int best = 0; float bv = -1e30f;
      for (int i = 0; i < Eh; ++i)
        if (!taken[i] && ms[i] > bv) { best = i; bv = ms[i]; }
      taken[best] = true; ind[j] = best; wv[j] = orig[best]; wsum += wv[j];
    }
    const float inv = 2.5f / (wsum + 1e-20f);
    for (int j = 0; j < TOPKh; ++j) {
      inds[t * TOPKh + j] = ind[j];
      wts[t * TOPKh + j]  = wv[j] * inv;
    }
  }
}

// ---------------------------------------------------------------------------
// Per-expert token lists: counts -> exclusive scan -> scatter (single block).
// ---------------------------------------------------------------------------
__global__ __launch_bounds__(256)
void permute_kernel(const int* __restrict__ inds,
                    const float* __restrict__ wts,
                    int* __restrict__ offsets,
                    int* __restrict__ ptok,
                    float* __restrict__ pwt) {
  __shared__ int cnt[Eh];
  __shared__ int cur[Eh + 1];
  const int tid = threadIdx.x;
  if (tid < Eh) cnt[tid] = 0;
  __syncthreads();
  for (int i = tid; i < Th * TOPKh; i += 256) atomicAdd(&cnt[inds[i]], 1);
  __syncthreads();
  if (tid == 0) {
    int s = 0;
    for (int e = 0; e <= Eh; ++e) { cur[e] = s; if (e < Eh) s += cnt[e]; }
  }
  __syncthreads();
  if (tid <= Eh) offsets[tid] = cur[tid];
  __syncthreads();
  for (int i = tid; i < Th * TOPKh; i += 256) {
    const int e = inds[i];
    const int pos = atomicAdd(&cur[e], 1);
    ptok[pos] = i >> 2;
    pwt[pos]  = wts[i];
  }
}

// ---------------------------------------------------------------------------
// Fused expert FFN: out[tile] (+)= coeff * (silu(X Wg^T) * (X Wu^T)) Wd^T
// ---------------------------------------------------------------------------
template <bool IS_SHARED>
__global__ __launch_bounds__(256)
void moe_expert_kernel(const float* __restrict__ x,
                       const float* __restrict__ Wg,
                       const float* __restrict__ Wu,
                       const float* __restrict__ Wd,
                       const int* __restrict__ offsets,
                       const int* __restrict__ ptok,
                       const float* __restrict__ pwt,
                       float* __restrict__ out) {
  extern __shared__ char smem[];
  __bf16* Hs  = (__bf16*)smem;                       // [TMh][DFFh]
  float*  F0  = (float*)(smem + HS_BYTES);           // fp32 stage buf 0
  float*  F1  = F0 + 2 * 64 * KSLAB;                 // fp32 stage buf 1
  __bf16* B0  = (__bf16*)(smem + HS_BYTES + F_BYTES);// bf16 stage buf 0
  __bf16* B1  = B0 + 2 * 64 * KSLAB;                 // bf16 stage buf 1
  __shared__ int   s_tok[TMh];
  __shared__ float s_cw[TMh];
  (void)F0; (void)F1;

  int e = 0, base = 0, end = 0;
  if (!IS_SHARED) {
    e = blockIdx.y;
    base = offsets[e];
    end  = offsets[e + 1];
    if (base + (int)blockIdx.x * TMh >= end) return;   // empty tile
  }
  const size_t wgo = (size_t)e * DFFh * Dh;
  const size_t wdo = (size_t)e * Dh * DFFh;

  const int tid = threadIdx.x;
  for (int i = tid; i < TMh; i += 256) {
    if (IS_SHARED) {
      s_tok[i] = blockIdx.x * TMh + i;
      s_cw[i]  = 1.0f;
    } else {
      const int slot = base + blockIdx.x * TMh + i;
      if (slot < end) { s_tok[i] = ptok[slot]; s_cw[i] = pwt[slot]; }
      else            { s_tok[i] = -1;         s_cw[i] = 0.0f;      }
    }
  }
  __syncthreads();

  const int wave = tid >> 5;     // M-subtile owner
  const int lane = tid & 31;
  const int idx  = lane & 15;
  const int half = lane >> 4;

  const int mrow = wave * 16 + idx;
  int tokA = s_tok[mrow];
  if (tokA < 0) tokA = 0;
  const float* __restrict__ xrow = x + (size_t)tokA * Dh;

  // ---------------- Phase 1: H = silu(X Wg^T) * (X Wu^T) ----------------
  for (int nf = 0; nf < DFFh; nf += 64) {
    const float* g0 = Wg + wgo + (size_t)nf * Dh;
    const float* u0 = Wu + wgo + (size_t)nf * Dh;
    FragAcc aG[4], aU[4];
#pragma unroll
    for (int q = 0; q < 4; ++q)
#pragma unroll
      for (int r = 0; r < 8; ++r) { aG[q].f[r] = 0.f; aU[q].f[r] = 0.f; }

    const int S = Dh / KSLAB;   // 64 slabs
#if USE_ASYNC
    stage_copy_async<2>(F0, g0, u0, Dh, 0, tid);
    for (int s = 0; s < S; ++s) {
      const int kb = s * KSLAB;
      __builtin_amdgcn_s_wait_asynccnt(0);   // slab s landed (this wave's copies)
      __syncthreads();                       // ...visible to every wave
      if (s + 1 < S)
        stage_copy_async<2>((s & 1) ? F0 : F1, g0, u0, Dh, kb + KSLAB, tid);
      stage_convert<2>((s & 1) ? F1 : F0, (s & 1) ? B1 : B0, tid);
      __syncthreads();                       // bf16 slab ready
      const __bf16* Bc = (s & 1) ? B1 : B0;
#else
    stage_fill_sync<2>(B0, g0, u0, Dh, 0, tid);
    for (int s = 0; s < S; ++s) {
      const int kb = s * KSLAB;
      __syncthreads();                       // B[s&1] ready; B[(s+1)&1] free
      if (s + 1 < S)
        stage_fill_sync<2>((s & 1) ? B0 : B1, g0, u0, Dh, kb + KSLAB, tid);
      const __bf16* Bc = (s & 1) ? B1 : B0;
#endif
      const v16bf A = make_bfrag(xrow + kb + half * 8, xrow + kb + 16 + half * 8);
#pragma unroll
      for (int q = 0; q < 4; ++q) {
        const int row = q * 16 + idx;
        const v16bf Bg = bfrag_stage(Bc, 0, row, half);
        const v16bf Bu = bfrag_stage(Bc, 1, row, half);
        aG[q].v = __builtin_amdgcn_wmma_f32_16x16x32_bf16(
            false, A, false, Bg, (short)0, aG[q].v, false, false);
        aU[q].v = __builtin_amdgcn_wmma_f32_16x16x32_bf16(
            false, A, false, Bu, (short)0, aU[q].v, false, false);
      }
    }
    // silu(gate) * up -> LDS H (bf16); D-matrix layout: m = r + 8*half
#pragma unroll
    for (int q = 0; q < 4; ++q) {
      const int col = nf + q * 16 + idx;
#pragma unroll
      for (int r = 0; r < 8; ++r) {
        const int m = wave * 16 + r + half * 8;
        const float g = aG[q].f[r];
        const float h = g * fast_sigmoid(g) * aU[q].f[r];
        Hs[(size_t)m * DFFh + col] = (__bf16)h;
      }
    }
  }
  __syncthreads();   // phase boundary (staging buffers are reused)

  // ---------------- Phase 2: Out = H Wd^T, scatter-accumulate ------------
  for (int nd = 0; nd < Dh; nd += 64) {
    const float* d0 = Wd + wdo + (size_t)nd * DFFh;
    FragAcc acc[4];
#pragma unroll
    for (int q = 0; q < 4; ++q)
#pragma unroll
      for (int r = 0; r < 8; ++r) acc[q].f[r] = 0.f;

    const int S = DFFh / KSLAB;   // 32 slabs
#if USE_ASYNC
    stage_copy_async<1>(F0, d0, d0, DFFh, 0, tid);
    for (int s = 0; s < S; ++s) {
      const int kb = s * KSLAB;
      __builtin_amdgcn_s_wait_asynccnt(0);
      __syncthreads();
      if (s + 1 < S)
        stage_copy_async<1>((s & 1) ? F0 : F1, d0, d0, DFFh, kb + KSLAB, tid);
      stage_convert<1>((s & 1) ? F1 : F0, (s & 1) ? B1 : B0, tid);
      __syncthreads();
      const __bf16* Bc = (s & 1) ? B1 : B0;
#else
    stage_fill_sync<1>(B0, d0, d0, DFFh, 0, tid);
    for (int s = 0; s < S; ++s) {
      const int kb = s * KSLAB;
      __syncthreads();
      if (s + 1 < S)
        stage_fill_sync<1>((s & 1) ? B0 : B1, d0, d0, DFFh, kb + KSLAB, tid);
      const __bf16* Bc = (s & 1) ? B1 : B0;
#endif
      FragBF fa;   // A-frag from own H rows: two ds_load_b128
      fa.q[0] = *(const u32x4*)(Hs + (size_t)mrow * DFFh + kb + half * 8);
      fa.q[1] = *(const u32x4*)(Hs + (size_t)mrow * DFFh + kb + 16 + half * 8);
#pragma unroll
      for (int q = 0; q < 4; ++q) {
        const v16bf Bd = bfrag_stage(Bc, 0, q * 16 + idx, half);
        acc[q].v = __builtin_amdgcn_wmma_f32_16x16x32_bf16(
            false, fa.v, false, Bd, (short)0, acc[q].v, false, false);
      }
    }
#pragma unroll
    for (int q = 0; q < 4; ++q) {
      const int col = nd + q * 16 + idx;
#pragma unroll
      for (int r = 0; r < 8; ++r) {
        const int m = wave * 16 + r + half * 8;
        const int tok = s_tok[m];
        if (IS_SHARED) {
          out[(size_t)tok * Dh + col] = acc[q].f[r];
        } else if (tok >= 0) {
          atomicAdd(out + (size_t)tok * Dh + col, s_cw[m] * acc[q].f[r]);
        }
      }
    }
    __syncthreads();   // staging buffers recycled next nd iteration
  }
}

// ---------------------------------------------------------------------------
// Launch
// ---------------------------------------------------------------------------
extern "C" void kernel_launch(void* const* d_in, const int* in_sizes, int n_in,
                              void* d_out, int out_size, void* d_ws, size_t ws_size,
                              hipStream_t stream) {
  const float* x      = (const float*)d_in[0];
  const float* gate_w = (const float*)d_in[1];
  const float* bias   = (const float*)d_in[2];
  const float* Wg     = (const float*)d_in[3];
  const float* Wu     = (const float*)d_in[4];
  const float* Wd     = (const float*)d_in[5];
  const float* sWg    = (const float*)d_in[6];
  const float* sWu    = (const float*)d_in[7];
  const float* sWd    = (const float*)d_in[8];
  float* out = (float*)d_out;

  char* ws = (char*)d_ws;
  int*   inds = (int*)(ws + 0);
  float* wts  = (float*)(ws + 16384);
  int*   offs = (int*)(ws + 32768);
  int*   ptok = (int*)(ws + 33280);
  float* pwt  = (float*)(ws + 49664);

  router_kernel<<<Th, 256, 0, stream>>>(x, gate_w, bias, inds, wts);
  permute_kernel<<<1, 256, 0, stream>>>(inds, wts, offs, ptok, pwt);

  // Shared expert first: plain stores initialize d_out.
  moe_expert_kernel<true><<<Th / TMh, 256, LDS_TOTAL, stream>>>(
      x, sWg, sWu, sWd, nullptr, nullptr, nullptr, out);
  // Routed experts: atomic accumulate with routing coefficients.
  moe_expert_kernel<false><<<dim3(Th / TMh, Eh), 256, LDS_TOTAL, stream>>>(
      x, Wg, Wu, Wd, offs, ptok, pwt, out);
}
